// GraphEncoder_51788715655836
// MI455X (gfx1250) — matrix operand order repaired and checked by reading, not compiled, not run on hardware
//
#include <hip/hip_runtime.h>

typedef __attribute__((ext_vector_type(16))) __bf16 v16bf;
typedef __attribute__((ext_vector_type(2)))  __bf16 v2bf;
typedef __attribute__((ext_vector_type(8)))  float  v8f;
typedef __attribute__((ext_vector_type(4)))  float  v4f;
typedef __attribute__((ext_vector_type(4)))  int    v4i;
typedef __attribute__((ext_vector_type(4)))  unsigned int v4u;

#define GLOBAL_AS __attribute__((address_space(1)))
typedef const GLOBAL_AS v4u* gv4u_p;
typedef const GLOBAL_AS v4i* gv4i_p;
typedef const GLOBAL_AS v4f* gv4f_p;

#define GAT_ALPHA    0.2f
#define GAT_LOG2E    1.4426950408889634f
// -9e15 * log2(e): masked-score constant already in the base-2 domain
#define GAT_NEG_INF2 -1.2984256e16f

// fp32 -> bf16 through the native conversion (v_cvt_*bf16_f32 when available)
__device__ __forceinline__ unsigned short f2bf(float f) {
  union { __bf16 b; unsigned short s; } cv;
  cv.b = (__bf16)f;
  return cv.s;
}

__device__ __forceinline__ float fexp2(float x) {
#if __has_builtin(__builtin_amdgcn_exp2f)
  return __builtin_amdgcn_exp2f(x);
#else
  return exp2f(x);
#endif
}

union PackAB { v16bf v; v2bf h[8]; unsigned short u[16]; v4u w[2]; };

__device__ __forceinline__ void pkpair(PackAB& P, int p, float f0, float f1) {
#if __has_builtin(__builtin_amdgcn_cvt_pk_bf16_f32)
  P.h[p] = __builtin_amdgcn_cvt_pk_bf16_f32(f0, f1);
#else
  P.v[2 * p]     = (__bf16)f0;
  P.v[2 * p + 1] = (__bf16)f1;
#endif
}

// ---------------------------------------------------------------------------
// Panel layout for bf16 B-operands:
//   matrix [K rows][Ncols] row-major  ->  panels nt = col/16, kb = row/32.
//   Each panel block = 512 ushorts (1 KB): [lane' = (row%32)/16*16 + col%16][kloc = row%16]
//   A lane's whole 32x16 WMMA B-fragment = 16 contiguous ushorts at lane'*16.
// ---------------------------------------------------------------------------

// Pre-swizzle W (fp32 row-major DxD) into bf16 panel layout. One-shot, tiny.
__global__ __launch_bounds__(256)
void gat_prep_w_kernel(const float* __restrict__ W, unsigned short* __restrict__ Wp,
                       int D) {
  int idx   = blockIdx.x * 256 + threadIdx.x;     // over D*D
  int kloc  = idx & 15;
  int lanep = (idx >> 4) & 31;
  int blk   = idx >> 9;
  int kbn   = D >> 5;
  int kb    = blk % kbn;
  int nt    = blk / kbn;
  int krow  = kb * 32 + (lanep >> 4) * 16 + kloc;
  int col   = nt * 16 + (lanep & 15);
  Wp[idx] = f2bf(W[(size_t)krow * D + col]);
}

// ---------------------------------------------------------------------------
// Kernel 1: Wh = X @ W (WMMA bf16), writes Wh in panel layout, fuses the
// attention logits s_src = Wh.a_src, s_dst = Wh.a_dst from fp32 accumulators.
// ---------------------------------------------------------------------------
__global__ __launch_bounds__(128)
void gat_xw_kernel(const float* __restrict__ X,
                   const unsigned short* __restrict__ Wp,
                   const float* __restrict__ a_src, const float* __restrict__ a_dst,
                   unsigned short* __restrict__ WhP,
                   float* __restrict__ s_src, float* __restrict__ s_dst,
                   int N, int D) {
  __shared__ __align__(16) float sL[16];
  __shared__ __align__(16) float dL[16];

  const int tid  = threadIdx.x;
  const int lane = tid & 31;
  const int wave = tid >> 5;
  const int hi   = lane >> 4;
  const int lm   = lane & 15;
  const int i0   = blockIdx.x * 16;
  const int kbn  = D >> 5;

  if (tid < 16) { sL[tid] = 0.f; dL[tid] = 0.f; }
  __syncthreads();

  // live B-panel pointers (explicit global AS), advanced 1 KB per k-step
  gv4u_p bp[4];
#pragma unroll
  for (int t = 0; t < 4; ++t)
    bp[t] = (gv4u_p)(Wp + (((size_t)((wave * 4 + t) * kbn)) << 9) + lane * 16);

  v8f acc[4] = {};

  for (int kb = 0; kb < kbn; ++kb) {
    // --- batch all four B fragments first (latency hidden by A conversion) ---
    PackAB B[4];
#pragma unroll
    for (int t = 0; t < 4; ++t) {
      B[t].w[0] = bp[t][0];
      B[t].w[1] = bp[t][1];
      bp[t] += 64;                                   // 512 ushorts = 64 v4u
    }
    // --- A fragment: row (i0+lm), one contiguous 16-float run ---
    gv4f_p xr = (gv4f_p)(X + (size_t)(i0 + lm) * D + kb * 32 + hi * 16);
    v4f a0 = xr[0];
    v4f a1 = xr[1];
    v4f a2 = xr[2];
    v4f a3 = xr[3];
    PackAB A;
    pkpair(A, 0, a0.x, a0.y); pkpair(A, 1, a0.z, a0.w);
    pkpair(A, 2, a1.x, a1.y); pkpair(A, 3, a1.z, a1.w);
    pkpair(A, 4, a2.x, a2.y); pkpair(A, 5, a2.z, a2.w);
    pkpair(A, 6, a3.x, a3.y); pkpair(A, 7, a3.z, a3.w);
#pragma unroll
    for (int t = 0; t < 4; ++t)
      acc[t] = __builtin_amdgcn_wmma_f32_16x16x32_bf16(
          false, A.v, false, B[t].v, (short)0, acc[t], false, false);
  }

  // --- store Wh (bf16, panel layout) ---
  const int jbn = N >> 5;
#pragma unroll
  for (int t = 0; t < 4; ++t) {
    const int nt = wave * 4 + t;
#pragma unroll
    for (int r = 0; r < 8; ++r) {
      int j = i0 + r + hi * 8;                       // global row
      size_t off = (((size_t)(nt * jbn + (j >> 5))) << 9)
                 + (size_t)(((j >> 4) & 1) * 16 + lm) * 16 + (j & 15);
      WhP[off] = f2bf(acc[t][r]);
    }
  }

  // --- fused logits: per-row dot with a_src / a_dst ---
  float avs[4], avd[4];
#pragma unroll
  for (int t = 0; t < 4; ++t) {
    int c = wave * 64 + t * 16 + lm;
    avs[t] = a_src[c];
    avd[t] = a_dst[c];
  }
#pragma unroll
  for (int r = 0; r < 8; ++r) {
    float ps = acc[0][r] * avs[0] + acc[1][r] * avs[1]
             + acc[2][r] * avs[2] + acc[3][r] * avs[3];
    float pd = acc[0][r] * avd[0] + acc[1][r] * avd[1]
             + acc[2][r] * avd[2] + acc[3][r] * avd[3];
#pragma unroll
    for (int off = 8; off > 0; off >>= 1) {          // reduce within 16-lane half
      ps += __shfl_xor(ps, off, 32);
      pd += __shfl_xor(pd, off, 32);
    }
    if (lm == 0) {                                   // 2 lanes/wave, LDS float atomics
      atomicAdd(&sL[r + hi * 8], ps);
      atomicAdd(&dL[r + hi * 8], pd);
    }
  }
  __syncthreads();
  if (tid < 16) {
    s_src[i0 + tid] = sL[tid];
    s_dst[i0 + tid] = dL[tid];
  }
}

// ---------------------------------------------------------------------------
// Kernel 2: fused masked-softmax attention, out = relu(attn @ Wh).
// Flash-style online softmax in the base-2 domain (raw v_exp_f32);
// adjacency stream software-pipelined one iteration ahead + prefetched 8
// ahead; B fragments batched before the softmax math; rowM ping-ponged.
// ---------------------------------------------------------------------------
__global__ __launch_bounds__(128)
void gat_attn_kernel(const unsigned short* __restrict__ WhP,
                     const float* __restrict__ s_src,
                     const float* __restrict__ s_dst,
                     const int* __restrict__ adj,
                     float* __restrict__ out, int N, int D) {
  __shared__ __align__(16) float pT[16 * 32];        // scores * log2(e)
  __shared__ __align__(16) float rowM[2][16];        // ping-pong running max
  __shared__ __align__(16) float rowL[16];
  __shared__ __align__(16) float srcR[16];

  const int tid  = threadIdx.x;
  const int lane = tid & 31;
  const int wave = tid >> 5;
  const int hi   = lane >> 4;
  const int lm   = lane & 15;
  const int i0   = blockIdx.x * 16;
  const int jbn  = N >> 5;

  if (tid < 16) {
    srcR[tid]    = s_src[i0 + tid];
    rowM[0][tid] = -3.0e38f;
    rowL[tid]    = 0.f;
  }
  v8f acc[4] = {};

  const int arow  = tid >> 3;                        // phase-A row 0..15
  const int acolb = (tid & 7) * 4;                   // phase-A col base
  const int* adjr = adj + (size_t)(i0 + arow) * N + acolb;   // for prefetch
  gv4i_p avp = (gv4i_p)adjr;
  gv4f_p dvp = (gv4f_p)(s_dst + acolb);

  // live B-panel pointers (explicit global AS), advanced 1 KB per j-step
  gv4u_p bp[4];
#pragma unroll
  for (int t = 0; t < 4; ++t)
    bp[t] = (gv4u_p)(WhP + (((size_t)((wave * 4 + t) * jbn)) << 9) + lane * 16);

  // pipeline prologue: first adjacency + dst-logit vectors
  v4i av = avp[0];
  v4f dv = dvp[0];

  __syncthreads();

  for (int j0 = 0; j0 < N; j0 += 32) {
    const int cur = (j0 >> 5) & 1;
    // ---------- phase A: masked leaky-relu scores (x log2e) -> pT ----------
    {
      float si = srcR[arow];
      float e0 = si + dv.x;
      float e1 = si + dv.y;
      float e2 = si + dv.z;
      float e3 = si + dv.w;
      e0 = e0 > 0.f ? e0 : GAT_ALPHA * e0;
      e1 = e1 > 0.f ? e1 : GAT_ALPHA * e1;
      e2 = e2 > 0.f ? e2 : GAT_ALPHA * e2;
      e3 = e3 > 0.f ? e3 : GAT_ALPHA * e3;
      v4f ev;
      ev.x = av.x ? e0 * GAT_LOG2E : GAT_NEG_INF2;
      ev.y = av.y ? e1 * GAT_LOG2E : GAT_NEG_INF2;
      ev.z = av.z ? e2 * GAT_LOG2E : GAT_NEG_INF2;
      ev.w = av.w ? e3 * GAT_LOG2E : GAT_NEG_INF2;
      *(v4f*)(pT + arow * 32 + acolb) = ev;
      // issue next iteration's streams now; consumed after the barriers
      if (j0 + 32 < N) {
        av = avp[(j0 + 32) >> 2];
        dv = dvp[(j0 + 32) >> 2];
      }
      if (j0 + 256 < N)
        __builtin_prefetch((const void*)(adjr + j0 + 256), 0, 1);
    }
    __syncthreads();

    // ---------- phase C ----------
    // batch all four B fragments first; softmax math hides their latency
    PackAB B[4];
#pragma unroll
    for (int t = 0; t < 4; ++t) {
      B[t].w[0] = bp[t][0];
      B[t].w[1] = bp[t][1];
      bp[t] += 64;                                   // 512 ushorts = 64 v4u
    }

    float mold = rowM[cur][lm];
    const float* pr = pT + lm * 32 + hi * 16;        // lane's A-layout runs
    v4f p0 = *(const v4f*)(pr + 0);
    v4f p1 = *(const v4f*)(pr + 4);
    v4f p2 = *(const v4f*)(pr + 8);
    v4f p3 = *(const v4f*)(pr + 12);
    float e[16] = { p0.x, p0.y, p0.z, p0.w, p1.x, p1.y, p1.z, p1.w,
                    p2.x, p2.y, p2.z, p2.w, p3.x, p3.y, p3.z, p3.w };
    float mx = mold;
#pragma unroll
    for (int q = 0; q < 16; ++q) mx = fmaxf(mx, e[q]);
    mx = fmaxf(mx, __shfl_xor(mx, 16, 32));          // lanes m / m+16 = full row
    float sc = fexp2(mold - mx);
    float pe[16];
    float lsum = 0.f;
#pragma unroll
    for (int q = 0; q < 16; ++q) {
      pe[q] = fexp2(e[q] - mx);
      lsum += pe[q];
    }
    PackAB A;
#pragma unroll
    for (int p = 0; p < 8; ++p) pkpair(A, p, pe[2 * p], pe[2 * p + 1]);
    lsum += __shfl_xor(lsum, 16, 32);

    // per-output-row scales for the C-layout accumulators
    float scr[8];
#pragma unroll
    for (int r = 0; r < 8; ++r) scr[r] = __shfl(sc, r + hi * 8, 32);

#pragma unroll
    for (int t = 0; t < 4; ++t) {
#pragma unroll
      for (int r = 0; r < 8; ++r) acc[t][r] *= scr[r];
      acc[t] = __builtin_amdgcn_wmma_f32_16x16x32_bf16(
          false, A.v, false, B[t].v, (short)0, acc[t], false, false);
    }

    if (wave == 0 && lane < 16) {                    // single writer per row
      rowM[cur ^ 1][lane] = mx;                      // ping-pong: no extra barrier
      rowL[lane] = rowL[lane] * sc + lsum;
    }
    __syncthreads();                                 // pT reuse + rowM visibility
  }

  // ---------- normalize, ReLU, store fp32 ----------
#pragma unroll
  for (int t = 0; t < 4; ++t) {
    int n0 = wave * 64 + t * 16;
#pragma unroll
    for (int r = 0; r < 8; ++r) {
      float l = rowL[r + hi * 8];
      float v = (l > 0.f) ? acc[t][r] / l : 0.f;
      out[(size_t)(i0 + r + hi * 8) * D + n0 + lm] = v > 0.f ? v : 0.f;
    }
  }
}

// ---------------------------------------------------------------------------
extern "C" void kernel_launch(void* const* d_in, const int* in_sizes, int n_in,
                              void* d_out, int out_size, void* d_ws, size_t ws_size,
                              hipStream_t stream) {
  const float* x   = (const float*)d_in[0];
  const int*   adj = (const int*)  d_in[1];
  const float* W1  = (const float*)d_in[2];
  const float* a1s = (const float*)d_in[3];
  const float* a1d = (const float*)d_in[4];
  const float* W2  = (const float*)d_in[5];
  const float* a2s = (const float*)d_in[6];
  const float* a2d = (const float*)d_in[7];

  const int D = in_sizes[3];          // 256
  const int N = in_sizes[0] / D;      // 8192

  // workspace: WhP bf16 | s_src | s_dst | h1 fp32 | Wp1 bf16 | Wp2 bf16
  char* ws = (char*)d_ws;
  unsigned short* whp = (unsigned short*)ws;
  float* ssrc = (float*)(ws + (size_t)N * D * 2);
  float* sdst = ssrc + N;
  float* h1   = (float*)(ws + (size_t)N * D * 2 + (size_t)2 * N * 4);
  unsigned short* wp1 = (unsigned short*)((char*)h1 + (size_t)N * D * 4);
  unsigned short* wp2 = wp1 + (size_t)D * D;

  const int rb = N / 16;              // 512 row blocks
  dim3 b128(128), b256(256);

  gat_prep_w_kernel<<<D * D / 256, b256, 0, stream>>>(W1, wp1, D);
  gat_prep_w_kernel<<<D * D / 256, b256, 0, stream>>>(W2, wp2, D);

  // ----- layer 1 -----
  gat_xw_kernel  <<<rb, b128, 0, stream>>>(x,  wp1, a1s, a1d, whp, ssrc, sdst, N, D);
  gat_attn_kernel<<<rb, b128, 0, stream>>>(whp, ssrc, sdst, adj, h1, N, D);

  // ----- layer 2 -----
  gat_xw_kernel  <<<rb, b128, 0, stream>>>(h1, wp2, a2s, a2d, whp, ssrc, sdst, N, D);
  gat_attn_kernel<<<rb, b128, 0, stream>>>(whp, ssrc, sdst, adj, (float*)d_out, N, D);
}
